// SelfAttention_40484361732754
// MI455X (gfx1250) — matrix operand order repaired and checked by reading, not compiled, not run on hardware
//
#include <hip/hip_runtime.h>
#include <hip/hip_bf16.h>

typedef __attribute__((ext_vector_type(16))) __bf16 v16bf;
typedef __attribute__((ext_vector_type(8)))  float  v8f;

#define WMMA_BF16(a,b,c) __builtin_amdgcn_wmma_f32_16x16x32_bf16(false,(a),false,(b),(short)0,(c),false,false)

union Frag {
  v16bf v;
  uint4 q[2];
};

__device__ __forceinline__ unsigned short f2bf(float f) {
  unsigned int u = __float_as_uint(f);
  u = u + 0x7FFFu + ((u >> 16) & 1u);   // round-to-nearest-even
  return (unsigned short)(u >> 16);
}
__device__ __forceinline__ unsigned int pack2(float a, float b) {
  return (unsigned int)f2bf(a) | ((unsigned int)f2bf(b) << 16);
}

// CDNA5 async copy: 16 bytes global -> LDS per lane, tracked by ASYNCcnt.
__device__ __forceinline__ void async_ld16(unsigned short* lds_dst, const unsigned short* gsrc) {
  unsigned loff = (unsigned)(unsigned long long)(void*)lds_dst;  // low 32b of generic addr = LDS offset
  asm volatile("global_load_async_to_lds_b128 %0, %1, off"
               :: "v"(loff), "v"(gsrc) : "memory");
}

// Problem constants (fixed by reference)
constexpr int Bn = 4;      // batch
constexpr int Kd = 128;    // head dim / model dim
constexpr int Tn = 2048;   // sequence
constexpr int Hn = 8;      // heads
constexpr int Jn = 1024;   // Hn*Kd

// ---------------------------------------------------------------- cvt f32->bf16
__global__ void cvt_kernel(const float* __restrict__ src, unsigned short* __restrict__ dst, int n) {
  int i = blockIdx.x * blockDim.x + threadIdx.x;
  if (i < n) dst[i] = f2bf(src[i]);
}

// ------------------------------------------------- x [B][128][2048] -> XT [B][2048][128] (bf16)
__global__ void xpose_kernel(const float* __restrict__ x, unsigned short* __restrict__ XT) {
  __shared__ float tile[32][33];
  int bt = blockIdx.x;
  int tt = bt % (Tn / 32);
  int kt = (bt / (Tn / 32)) % (Kd / 32);
  int b  = bt / ((Tn / 32) * (Kd / 32));
  int tx = threadIdx.x;   // 0..31
  int ty = threadIdx.y;   // 0..7
  const float* xb = x + (size_t)b * Kd * Tn;
#pragma unroll
  for (int j = 0; j < 32; j += 8)
    tile[ty + j][tx] = xb[(size_t)(kt * 32 + ty + j) * Tn + tt * 32 + tx];
  __syncthreads();
  unsigned short* xt = XT + (size_t)b * Tn * Kd;
#pragma unroll
  for (int j = 0; j < 32; j += 8)
    xt[(size_t)(tt * 32 + ty + j) * Kd + kt * 32 + tx] = f2bf(tile[tx][ty + j]);
}

// ------------------------------------------------- P[j,t] = sum_k W[j,k] * X[k,t]
// W bf16 [1024][128], XT bf16 [B][2048][128]
// transposed==1: store P^T as [B][H][T][128]   (for Q, K)
// transposed==0: store P   as [B][H][128][T]   (for V)
__global__ void proj_kernel(const unsigned short* __restrict__ W,
                            const unsigned short* __restrict__ XT,
                            unsigned short* __restrict__ out, int transposed) {
  int wave = blockIdx.x * (blockDim.x >> 5) + (threadIdx.x >> 5);
  int lane = threadIdx.x & 31;
  int tn = wave % (Tn / 32);
  int tm = (wave / (Tn / 32)) % (Jn / 32);
  int b  = wave / ((Tn / 32) * (Jn / 32));
  int j0 = tm * 32, t0 = tn * 32;
  int lh = lane & 15;
  bool lo = lane < 16;

  Frag A[2][4];
#pragma unroll
  for (int mi = 0; mi < 2; mi++) {
    const unsigned short* p = W + (size_t)(j0 + 16 * mi + lh) * Kd;
#pragma unroll
    for (int ks = 0; ks < 4; ks++) {
      int kb = 32 * ks + (lo ? 0 : 16);
      A[mi][ks].q[0] = *(const uint4*)(p + kb);
      A[mi][ks].q[1] = *(const uint4*)(p + kb + 8);
    }
  }

  v8f acc[2][2] = {};
#pragma unroll
  for (int ks = 0; ks < 4; ks++) {
    Frag Bf[2];
#pragma unroll
    for (int ni = 0; ni < 2; ni++) {
      const unsigned short* p =
          XT + ((size_t)b * Tn + t0 + 16 * ni + lh) * Kd + 32 * ks + (lo ? 0 : 16);
      Bf[ni].q[0] = ((const uint4*)p)[0];
      Bf[ni].q[1] = ((const uint4*)p)[1];
    }
#pragma unroll
    for (int mi = 0; mi < 2; mi++)
#pragma unroll
      for (int ni = 0; ni < 2; ni++)
        acc[mi][ni] = WMMA_BF16(A[mi][ks].v, Bf[ni].v, acc[mi][ni]);
  }

#pragma unroll
  for (int mi = 0; mi < 2; mi++) {
    int jb = j0 + 16 * mi + (lo ? 0 : 8);
    int h  = jb >> 7;
    int jl = jb & 127;
#pragma unroll
    for (int ni = 0; ni < 2; ni++) {
      int t = t0 + 16 * ni + lh;
      if (transposed) {
        uint4 pk;
        pk.x = pack2(acc[mi][ni][0], acc[mi][ni][1]);
        pk.y = pack2(acc[mi][ni][2], acc[mi][ni][3]);
        pk.z = pack2(acc[mi][ni][4], acc[mi][ni][5]);
        pk.w = pack2(acc[mi][ni][6], acc[mi][ni][7]);
        *(uint4*)(out + (((size_t)(b * Hn + h) * Tn + t) * Kd + jl)) = pk;
      } else {
#pragma unroll
        for (int r = 0; r < 8; r++)
          out[((size_t)(b * Hn + h) * Kd + jl + r) * Tn + t] = f2bf(acc[mi][ni][r]);
      }
    }
  }
}

// ------------------------------------------------- flash attention, block-cooperative
// QT,KT: [B][H][T][128] bf16;  Vn: [B][H][128][T] bf16;  OT: [B][T][H*128] bf16
// Block = 8 waves on one (b,h,128-row t-block); K/V chunks double-buffered in LDS
// via global_load_async_to_lds_b128.
__device__ __forceinline__ void stage_chunk(const unsigned short* Kb, const unsigned short* Vb,
                                            unsigned short* Ksh, unsigned short* Vsh,
                                            int i0, int tid) {
  // K chunk: rows i0..i0+31 of [T][128] are contiguous -> flat 8KB copy
#pragma unroll
  for (int j = 0; j < 2; j++) {
    int idx = j * 256 + tid;                  // 0..511 chunks of 16B
    async_ld16(Ksh + idx * 8, Kb + (size_t)i0 * Kd + idx * 8);
  }
  // V chunk: [128][32] sub-block of [128][T]
#pragma unroll
  for (int j = 0; j < 2; j++) {
    int c = j * 256 + tid;                    // 0..511
    int r = c >> 2;                           // 0..127
    int cc = (c & 3) * 8;                     // 0,8,16,24
    async_ld16(Vsh + r * 32 + cc, Vb + (size_t)r * Tn + i0 + cc);
  }
}

__global__ void attn_kernel(const unsigned short* __restrict__ QT,
                            const unsigned short* __restrict__ KT,
                            const unsigned short* __restrict__ Vg,
                            unsigned short* __restrict__ OT) {
  __shared__ uint4 KldU[2][512];   // 2 x (32 x 128) bf16
  __shared__ uint4 VldU[2][512];   // 2 x (128 x 32) bf16
  __shared__ uint4 PldU[8][64];    // per-wave 16x32 bf16 prob tile

  int tid = threadIdx.x;
  int wslot = tid >> 5;
  int lane = tid & 31;
  int lh = lane & 15;
  bool lo = lane < 16;

  int blk = blockIdx.x;
  int tblk = blk % (Tn / 128);
  int h  = (blk / (Tn / 128)) % Hn;
  int b  = blk / ((Tn / 128) * Hn);
  int t0 = tblk * 128 + wslot * 16;

  const unsigned short* Qb = QT + (size_t)(b * Hn + h) * Tn * Kd;
  const unsigned short* Kb = KT + (size_t)(b * Hn + h) * Tn * Kd;
  const unsigned short* Vb = Vg + (size_t)(b * Hn + h) * Kd * Tn;

  unsigned short* Ksh0 = (unsigned short*)KldU[0];
  unsigned short* Ksh1 = (unsigned short*)KldU[1];
  unsigned short* Vsh0 = (unsigned short*)VldU[0];
  unsigned short* Vsh1 = (unsigned short*)VldU[1];
  unsigned short* P    = (unsigned short*)PldU[wslot];

  // prologue: stage chunk 0
  stage_chunk(Kb, Vb, Ksh0, Vsh0, 0, tid);

  Frag Aq[4];
  {
    const unsigned short* p = Qb + (size_t)(t0 + lh) * Kd;
#pragma unroll
    for (int ks = 0; ks < 4; ks++) {
      int kb = 32 * ks + (lo ? 0 : 16);
      Aq[ks].q[0] = *(const uint4*)(p + kb);
      Aq[ks].q[1] = *(const uint4*)(p + kb + 8);
    }
  }

  v8f O[8] = {};
  float m[8], l[8];
#pragma unroll
  for (int r = 0; r < 8; r++) { m[r] = -1e30f; l[r] = 0.f; }

  const float sc  = 0.08838834764831845f;  // 1/sqrt(128)
  const float L2E = 1.4426950408889634f;

  for (int i0 = 0; i0 < Tn; i0 += 32) {
    int cb = (i0 >> 5) & 1;
    unsigned short* Ksh = cb ? Ksh1 : Ksh0;
    unsigned short* Vsh = cb ? Vsh1 : Vsh0;
    if (i0 + 32 < Tn) {
      stage_chunk(Kb, Vb, cb ? Ksh0 : Ksh1, cb ? Vsh0 : Vsh1, i0 + 32, tid);
      // async loads complete in order: <=4 outstanding => current chunk landed
      asm volatile("s_wait_asynccnt 0x4" ::: "memory");
    } else {
      asm volatile("s_wait_asynccnt 0x0" ::: "memory");
    }
    __syncthreads();

    // ---- S(16x32) = Q^T K, K from LDS
    v8f S0 = {}, S1 = {};
#pragma unroll
    for (int ks = 0; ks < 4; ks++) {
      Frag B0, B1;
      const unsigned short* p0 = Ksh + (size_t)lh * Kd + 32 * ks + (lo ? 0 : 16);
      const unsigned short* p1 = Ksh + (size_t)(16 + lh) * Kd + 32 * ks + (lo ? 0 : 16);
      B0.q[0] = ((const uint4*)p0)[0]; B0.q[1] = ((const uint4*)p0)[1];
      B1.q[0] = ((const uint4*)p1)[0]; B1.q[1] = ((const uint4*)p1)[1];
      S0 = WMMA_BF16(Aq[ks].v, B0.v, S0);
      S1 = WMMA_BF16(Aq[ks].v, B1.v, S1);
    }

    // ---- online softmax (rows live in 16-lane halves: VGPR r -> row r / r+8)
    float p0[8], p1[8], alpha[8];
#pragma unroll
    for (int r = 0; r < 8; r++) {
      float v0 = S0[r] * sc, v1 = S1[r] * sc;
      float mx = fmaxf(v0, v1);
      mx = fmaxf(mx, __shfl_xor(mx, 8));
      mx = fmaxf(mx, __shfl_xor(mx, 4));
      mx = fmaxf(mx, __shfl_xor(mx, 2));
      mx = fmaxf(mx, __shfl_xor(mx, 1));
      float mn = fmaxf(m[r], mx);
      alpha[r] = exp2f((m[r] - mn) * L2E);
      p0[r] = exp2f((v0 - mn) * L2E);
      p1[r] = exp2f((v1 - mn) * L2E);
      float s = p0[r] + p1[r];
      s += __shfl_xor(s, 8);
      s += __shfl_xor(s, 4);
      s += __shfl_xor(s, 2);
      s += __shfl_xor(s, 1);
      l[r] = l[r] * alpha[r] + s;
      m[r] = mn;
    }
#pragma unroll
    for (int nt = 0; nt < 8; nt++)
#pragma unroll
      for (int r = 0; r < 8; r++)
        O[nt][r] *= alpha[r];

    // ---- re-layout probs D-format -> A-format through per-wave LDS tile
#pragma unroll
    for (int r = 0; r < 8; r++) {
      int row = lo ? r : (r + 8);
      P[row * 32 + lh]      = f2bf(p0[r]);
      P[row * 32 + 16 + lh] = f2bf(p1[r]);
    }
    asm volatile("s_wait_dscnt 0x0" ::: "memory");
    Frag Ap;
    {
      int kb = lo ? 0 : 16;
      const unsigned short* pp = P + lh * 32;
      Ap.q[0] = *(const uint4*)(pp + kb);
      Ap.q[1] = *(const uint4*)(pp + kb + 8);
    }
    asm volatile("" ::: "memory");

    // ---- O(16x128) += P(16x32) * V^T(32x128), V from LDS
#pragma unroll
    for (int nt = 0; nt < 8; nt++) {
      Frag Bv;
      const unsigned short* p = Vsh + (size_t)(nt * 16 + lh) * 32 + (lo ? 0 : 16);
      Bv.q[0] = ((const uint4*)p)[0]; Bv.q[1] = ((const uint4*)p)[1];
      O[nt] = WMMA_BF16(Ap.v, Bv.v, O[nt]);
    }
    __syncthreads();   // all waves done with this buffer before it is re-staged
  }

  // ---- normalize and store O^T into OT [B][T][H*128]
#pragma unroll
  for (int r = 0; r < 8; r++) l[r] = 1.0f / l[r];
#pragma unroll
  for (int nt = 0; nt < 8; nt++) {
    int kl = nt * 16 + lh;
#pragma unroll
    for (int r = 0; r < 8; r++) {
      int t = t0 + r + (lo ? 0 : 8);
      OT[((size_t)b * Tn + t) * Jn + h * Kd + kl] = f2bf(O[nt][r] * l[r]);
    }
  }
}

// ------------------------------------------------- out[b,kk,t] = Wu[kk,:] . OT[b,t,:] + bu[kk]
__global__ void unify_kernel(const unsigned short* __restrict__ Wu,
                             const unsigned short* __restrict__ OT,
                             const float* __restrict__ bu,
                             float* __restrict__ out) {
  int wave = blockIdx.x * (blockDim.x >> 5) + (threadIdx.x >> 5);
  int lane = threadIdx.x & 31;
  int tn = wave % (Tn / 32);
  int tm = (wave / (Tn / 32)) % (Kd / 32);
  int b  = wave / ((Tn / 32) * (Kd / 32));
  int k0 = tm * 32, t0 = tn * 32;
  int lh = lane & 15;
  bool lo = lane < 16;

  v8f acc[2][2] = {};
  for (int ks = 0; ks < Jn / 32; ks++) {
    Frag A0, A1, B0, B1;
    {
      const unsigned short* p = Wu + (size_t)(k0 + lh) * Jn + 32 * ks + (lo ? 0 : 16);
      A0.q[0] = *(const uint4*)p; A0.q[1] = *(const uint4*)(p + 8);
      p += (size_t)16 * Jn;
      A1.q[0] = *(const uint4*)p; A1.q[1] = *(const uint4*)(p + 8);
    }
    {
      const unsigned short* p = OT + ((size_t)b * Tn + t0 + lh) * Jn + 32 * ks + (lo ? 0 : 16);
      B0.q[0] = ((const uint4*)p)[0]; B0.q[1] = ((const uint4*)p)[1];
      p += (size_t)16 * Jn;
      B1.q[0] = ((const uint4*)p)[0]; B1.q[1] = ((const uint4*)p)[1];
    }
    acc[0][0] = WMMA_BF16(A0.v, B0.v, acc[0][0]);
    acc[0][1] = WMMA_BF16(A0.v, B1.v, acc[0][1]);
    acc[1][0] = WMMA_BF16(A1.v, B0.v, acc[1][0]);
    acc[1][1] = WMMA_BF16(A1.v, B1.v, acc[1][1]);
  }
#pragma unroll
  for (int mi = 0; mi < 2; mi++) {
    int kb = k0 + 16 * mi + (lo ? 0 : 8);
#pragma unroll
    for (int ni = 0; ni < 2; ni++) {
      int t = t0 + 16 * ni + lh;
#pragma unroll
      for (int r = 0; r < 8; r++) {
        int kk = kb + r;
        out[((size_t)b * Kd + kk) * Tn + t] = acc[mi][ni][r] + bu[kk];
      }
    }
  }
}

extern "C" void kernel_launch(void* const* d_in, const int* in_sizes, int n_in,
                              void* d_out, int out_size, void* d_ws, size_t ws_size,
                              hipStream_t stream) {
  const float* x   = (const float*)d_in[0];
  const float* Wqf = (const float*)d_in[1];
  const float* Wkf = (const float*)d_in[2];
  const float* Wvf = (const float*)d_in[3];
  const float* Wuf = (const float*)d_in[4];
  const float* bu  = (const float*)d_in[5];
  float* out = (float*)d_out;

  unsigned short* wq = (unsigned short*)d_ws;            // [1024][128]
  unsigned short* wk = wq + (size_t)Jn * Kd;
  unsigned short* wv = wk + (size_t)Jn * Kd;
  unsigned short* wu = wv + (size_t)Jn * Kd;             // [128][1024]
  unsigned short* XT = wu + (size_t)Kd * Jn;             // [B][T][128]
  unsigned short* QT = XT + (size_t)Bn * Tn * Kd;        // [B][H][T][128]
  unsigned short* KT = QT + (size_t)Bn * Hn * Tn * Kd;
  unsigned short* Vw = KT + (size_t)Bn * Hn * Tn * Kd;   // [B][H][128][T]
  unsigned short* OT = Vw + (size_t)Bn * Hn * Tn * Kd;   // [B][T][1024]

  cvt_kernel<<<512, 256, 0, stream>>>(Wqf, wq, Jn * Kd);
  cvt_kernel<<<512, 256, 0, stream>>>(Wkf, wk, Jn * Kd);
  cvt_kernel<<<512, 256, 0, stream>>>(Wvf, wv, Jn * Kd);
  cvt_kernel<<<512, 256, 0, stream>>>(Wuf, wu, Kd * Jn);

  xpose_kernel<<<Bn * (Kd / 32) * (Tn / 32), dim3(32, 8), 0, stream>>>(x, XT);

  proj_kernel<<<(Bn * (Jn / 32) * (Tn / 32)) / 8, 256, 0, stream>>>(wq, XT, QT, 1);
  proj_kernel<<<(Bn * (Jn / 32) * (Tn / 32)) / 8, 256, 0, stream>>>(wk, XT, KT, 1);
  proj_kernel<<<(Bn * (Jn / 32) * (Tn / 32)) / 8, 256, 0, stream>>>(wv, XT, Vw, 0);

  attn_kernel<<<Bn * Hn * (Tn / 128), 256, 0, stream>>>(QT, KT, Vw, OT);

  unify_kernel<<<(Bn * (Kd / 32) * (Tn / 32)) / 8, 256, 0, stream>>>(wu, OT, bu, out);
}
